// TGATLayer_1331439862430
// MI455X (gfx1250) — matrix operand order repaired and checked by reading, not compiled, not run on hardware
//
#include <hip/hip_runtime.h>

// ---------------------------------------------------------------------------
// TGAT layer, fused per-dst-node kernel for MI455X (gfx1250, wave32, WMMA).
//
// Heavy math: per node kh/vh = m(32x320) @ W(320x512)  -> done with
// v_wmma_f32_16x16x32_bf16, one wave per attention head (N-slice of 64).
// Weights are pre-converted fp32->bf16 and pre-swizzled into WMMA B-fragment
// order in d_ws so the inner loop does only aligned b128 loads.
// ---------------------------------------------------------------------------

#define N_SRC   16384
#define N_DST   8192
#define KNB     16
#define TNB     2
#define L_MSG   32          // K*T messages per node
#define NODE_IN 128
#define EDGE_IN 64
#define TIME_D  128
#define D_MODEL 320
#define N_HEAD  8
#define D_K     64
#define QKV     512         // N_HEAD*D_K
#define HID     128
#define CAT_IN  448         // 2*NODE_IN + TIME_D + EDGE_IN
#define LN_EPS  1e-5f

#define M_STRIDE 336        // bf16 stride for m rows (16B aligned, conflict-free)
#define KV_STRIDE 66        // bf16 stride for per-head kh/vh tiles
#define KSTEPS   10         // 320 / 32
#define NT_PER_WAVE 4       // 64 cols / 16
#define FRAG_ELEMS 512      // 32 lanes * 16 bf16 per B fragment
#define FRAGS_PER_MAT (N_HEAD * NT_PER_WAVE * KSTEPS)      // 320
#define ELEMS_PER_MAT (FRAGS_PER_MAT * FRAG_ELEMS)         // 163840

typedef __attribute__((ext_vector_type(8)))  unsigned short ushort8;
typedef __attribute__((ext_vector_type(16))) __bf16        v16bf;
typedef __attribute__((ext_vector_type(8)))  float         v8f;

union FragBF {
  v16bf   v;
  ushort8 h[2];
};

__device__ __forceinline__ unsigned short f32_to_bf16(float x) {
  union { float f; unsigned int u; } c; c.f = x;
  unsigned int u = c.u;
  u += 0x7FFFu + ((u >> 16) & 1u);     // round-to-nearest-even
  return (unsigned short)(u >> 16);
}
__device__ __forceinline__ float bf16_to_f32(unsigned short h) {
  union { unsigned int u; float f; } c; c.u = ((unsigned int)h) << 16;
  return c.f;
}

// ---------------------------------------------------------------------------
// Kernel 1: convert w_ks / w_vs (512x320 fp32) into bf16 B-fragments.
// Fragment f = ((head*4 + nt)*10 + kstep); within a fragment, lane L holds
// column j = head*64 + nt*16 + (L&15), K-run c = kstep*32 + (L>=16?16:0) + e.
// Output element index = f*512 + lane*16 + e  (32 contiguous bytes per lane).
// ---------------------------------------------------------------------------
__global__ void wmma_weight_swizzle(const float* __restrict__ w_ks,
                                    const float* __restrict__ w_vs,
                                    unsigned short* __restrict__ ws_bf) {
  int id = blockIdx.x * blockDim.x + threadIdx.x;
  if (id >= 2 * ELEMS_PER_MAT) return;
  int mat = (id >= ELEMS_PER_MAT) ? 1 : 0;
  int r   = id - mat * ELEMS_PER_MAT;
  int f      = r >> 9;          // /512
  int within = r & 511;
  int lane   = within >> 4;
  int e      = within & 15;
  int h     = f / (NT_PER_WAVE * KSTEPS);
  int rem   = f - h * (NT_PER_WAVE * KSTEPS);
  int nt    = rem / KSTEPS;
  int kstep = rem - nt * KSTEPS;
  int j = h * D_K + nt * 16 + (lane & 15);
  int c = kstep * 32 + ((lane >= 16) ? 16 : 0) + e;
  const float* src = mat ? w_vs : w_ks;
  ws_bf[id] = f32_to_bf16(src[j * D_MODEL + c]);
}

// ---------------------------------------------------------------------------
// Kernel 2: fused TGAT layer. grid = N_DST blocks of 256 threads (8 waves).
// ---------------------------------------------------------------------------
__global__ __launch_bounds__(256)
void tgat_fused(const float* __restrict__ node_features,
                const float* __restrict__ edge_features,
                const float* __restrict__ delta_t,
                const int*   __restrict__ neigh_idx,
                const float* __restrict__ basis_freq,
                const float* __restrict__ phase,
                const float* __restrict__ w_qs,
                const unsigned short* __restrict__ wsK,
                const unsigned short* __restrict__ wsV,
                const float* __restrict__ fc_w,
                const float* __restrict__ fc_b,
                const float* __restrict__ ln_g,
                const float* __restrict__ ln_b,
                const float* __restrict__ node_w,
                const float* __restrict__ node_b,
                float* __restrict__ out) {
  __shared__ __align__(16) unsigned short m_bf[L_MSG * M_STRIDE];       // 21.5 KB
  __shared__ __align__(16) unsigned short kv_bf[N_HEAD][L_MSG * KV_STRIDE]; // 33 KB
  __shared__ float q_sh[D_MODEL];
  __shared__ float qh_sh[QKV];
  __shared__ float attn_sh[N_HEAD * L_MSG];
  __shared__ float ctx_sh[QKV];
  __shared__ float x_sh[D_MODEL];
  __shared__ float red_sh[16];

  const int n    = blockIdx.x;
  const int tid  = threadIdx.x;
  const int lane = tid & 31;
  const int w    = tid >> 5;            // wave id == attention head
  const int l15  = lane & 15;
  const int hi   = (lane >= 16) ? 1 : 0;

  // ---- Stage 1: build m (bf16) and q (f32) in LDS -------------------------
  for (int idx = tid; idx < L_MSG * D_MODEL; idx += 256) {
    int l = idx / D_MODEL;
    int c = idx - l * D_MODEL;
    int k = l >> 1;
    int t = l & 1;
    float val;
    if (c < NODE_IN) {
      int src = neigh_idx[n * KNB + k];
      val = node_features[src * NODE_IN + c];
    } else if (c < NODE_IN + EDGE_IN) {
      val = edge_features[((n * KNB + k) * TNB + t) * EDGE_IN + (c - NODE_IN)];
    } else {
      int d = c - (NODE_IN + EDGE_IN);
      val = cosf(delta_t[(n * KNB + k) * TNB + t] * basis_freq[d] + phase[d]);
    }
    m_bf[l * M_STRIDE + c] = f32_to_bf16(val);
  }
  if (tid < D_MODEL) {
    float val;
    if (tid < NODE_IN)                 val = node_features[n * NODE_IN + tid];
    else if (tid < NODE_IN + EDGE_IN)  val = 0.0f;
    else                               val = cosf(phase[tid - (NODE_IN + EDGE_IN)]);
    q_sh[tid] = val;
  }
  __syncthreads();

  // ---- Stage 2: qh = q @ w_qs.T (512-dim GEMV, 2 outputs/thread) ----------
  for (int j = tid; j < QKV; j += 256) {
    const float* wr = w_qs + j * D_MODEL;
    float acc = 0.f;
    for (int c = 0; c < D_MODEL; ++c) acc += q_sh[c] * wr[c];
    qh_sh[j] = acc;
  }
  __syncthreads();

  // ---- Stages 3&5: kh / vh = m @ W_head.T via WMMA bf16 -------------------
  v8f acc[2][NT_PER_WAVE];
  auto run_proj = [&](const unsigned short* __restrict__ wmat) {
#pragma unroll
    for (int mt = 0; mt < 2; ++mt)
#pragma unroll
      for (int nt = 0; nt < NT_PER_WAVE; ++nt)
#pragma unroll
        for (int i = 0; i < 8; ++i) acc[mt][nt][i] = 0.f;

    for (int kstep = 0; kstep < KSTEPS; ++kstep) {
      FragBF a[2], b[NT_PER_WAVE];
      // A fragments from LDS: lanes 0-15 hold K {0..15}, lanes 16-31
      // hold K {16..31} of this 32-wide K slab (matches B swizzle).
#pragma unroll
      for (int mt = 0; mt < 2; ++mt) {
        int row   = mt * 16 + l15;
        int cbase = kstep * 32 + hi * 16;
        a[mt].h[0] = *(const ushort8*)&m_bf[row * M_STRIDE + cbase];
        a[mt].h[1] = *(const ushort8*)&m_bf[row * M_STRIDE + cbase + 8];
      }
      // B fragments: pre-swizzled, 32 contiguous bytes per lane.
#pragma unroll
      for (int nt = 0; nt < NT_PER_WAVE; ++nt) {
        const unsigned short* p =
            wmat + ((w * NT_PER_WAVE + nt) * KSTEPS + kstep) * FRAG_ELEMS + lane * 16;
        b[nt].h[0] = *(const ushort8*)p;
        b[nt].h[1] = *(const ushort8*)(p + 8);
      }
#pragma unroll
      for (int mt = 0; mt < 2; ++mt)
#pragma unroll
        for (int nt = 0; nt < NT_PER_WAVE; ++nt)
          acc[mt][nt] = __builtin_amdgcn_wmma_f32_16x16x32_bf16(
              false, a[mt].v, false, b[nt].v, (short)0, acc[mt][nt],
              false, false);
    }
    // D layout: lane -> col 16*nt+(lane&15); vgpr r -> row 16*mt + r + 8*hi
#pragma unroll
    for (int mt = 0; mt < 2; ++mt)
#pragma unroll
      for (int nt = 0; nt < NT_PER_WAVE; ++nt)
#pragma unroll
        for (int r = 0; r < 8; ++r) {
          int row = mt * 16 + hi * 8 + r;
          int col = nt * 16 + l15;
          kv_bf[w][row * KV_STRIDE + col] = f32_to_bf16(acc[mt][nt][r]);
        }
  };

  run_proj(wsK);   // kh for head w now in kv_bf[w]

  // ---- Stage 4: scores + softmax, one message l per lane ------------------
  {
    float s = 0.f;
    const float* qh_h = qh_sh + w * D_K;
    for (int d = 0; d < D_K; ++d)
      s += qh_h[d] * bf16_to_f32(kv_bf[w][lane * KV_STRIDE + d]);
    s *= 0.125f;                                  // 1/sqrt(D_K)
    float mx = s;
    for (int off = 16; off > 0; off >>= 1)
      mx = fmaxf(mx, __shfl_xor(mx, off, 32));
    float e = __expf(s - mx);
    float sum = e;
    for (int off = 16; off > 0; off >>= 1)
      sum += __shfl_xor(sum, off, 32);
    attn_sh[w * L_MSG + lane] = e / sum;
  }

  run_proj(wsV);   // vh for head w overwrites kv_bf[w] (wave-private, in-order)

  // ---- Stage 6: ctx[h] = attn @ vh, 2 output dims per lane ----------------
#pragma unroll
  for (int dd = 0; dd < 2; ++dd) {
    int d = lane + dd * 32;
    float acc_c = 0.f;
    for (int l = 0; l < L_MSG; ++l)
      acc_c += attn_sh[w * L_MSG + l] * bf16_to_f32(kv_bf[w][l * KV_STRIDE + d]);
    ctx_sh[w * D_K + d] = acc_c;
  }
  __syncthreads();

  // ---- Stage 7: out = ctx @ fc_w.T + fc_b ; x = out + q -------------------
  for (int c = tid; c < D_MODEL; c += 256) {
    const float* wr = fc_w + c * QKV;
    float acc_f = fc_b[c];
    for (int j = 0; j < QKV; ++j) acc_f += ctx_sh[j] * wr[j];
    x_sh[c] = acc_f + q_sh[c];
  }
  __syncthreads();

  // ---- Stage 8: LayerNorm over 320 ---------------------------------------
  {
    float ls = 0.f, lq = 0.f;
    for (int c = tid; c < D_MODEL; c += 256) {
      float v = x_sh[c];
      ls += v; lq += v * v;
    }
    for (int off = 16; off > 0; off >>= 1) {
      ls += __shfl_xor(ls, off, 32);
      lq += __shfl_xor(lq, off, 32);
    }
    if (lane == 0) { red_sh[w] = ls; red_sh[8 + w] = lq; }
  }
  __syncthreads();
  {
    float s0 = 0.f, s1 = 0.f;
    for (int i = 0; i < 8; ++i) { s0 += red_sh[i]; s1 += red_sh[8 + i]; }
    float mu   = s0 / (float)D_MODEL;
    float var  = s1 / (float)D_MODEL - mu * mu;
    float rstd = rsqrtf(var + LN_EPS);
    for (int c = tid; c < D_MODEL; c += 256)
      x_sh[c] = ln_g[c] * (x_sh[c] - mu) * rstd + ln_b[c];
  }
  __syncthreads();

  // ---- Stage 9: relu([h_dst, h_neigh] @ node_w.T + node_b) ----------------
  for (int o = tid; o < HID; o += 256) {
    const float* wr = node_w + o * CAT_IN;
    float acc_o = node_b[o];
    for (int c = 0; c < NODE_IN; ++c) acc_o += q_sh[c] * wr[c];           // h_dst
    for (int c = 0; c < D_MODEL; ++c) acc_o += x_sh[c] * wr[NODE_IN + c]; // h_neigh
    out[n * HID + o] = fmaxf(acc_o, 0.f);
  }
}

// ---------------------------------------------------------------------------
extern "C" void kernel_launch(void* const* d_in, const int* in_sizes, int n_in,
                              void* d_out, int out_size, void* d_ws, size_t ws_size,
                              hipStream_t stream) {
  (void)in_sizes; (void)n_in; (void)out_size; (void)ws_size;
  const float* node_features = (const float*)d_in[0];
  const float* edge_features = (const float*)d_in[1];
  const float* delta_t       = (const float*)d_in[2];
  const int*   neigh_idx     = (const int*)d_in[3];
  const float* basis_freq    = (const float*)d_in[4];
  const float* phase         = (const float*)d_in[5];
  const float* w_qs          = (const float*)d_in[6];
  const float* w_ks          = (const float*)d_in[7];
  const float* w_vs          = (const float*)d_in[8];
  const float* fc_w          = (const float*)d_in[9];
  const float* fc_b          = (const float*)d_in[10];
  const float* ln_g          = (const float*)d_in[11];
  const float* ln_b          = (const float*)d_in[12];
  const float* node_w        = (const float*)d_in[13];
  const float* node_b        = (const float*)d_in[14];

  // Workspace: 2 * 163840 bf16 = 640 KB of pre-swizzled K/V weights.
  unsigned short* ws_bf = (unsigned short*)d_ws;
  int tot = 2 * ELEMS_PER_MAT;
  wmma_weight_swizzle<<<(tot + 255) / 256, 256, 0, stream>>>(w_ks, w_vs, ws_bf);

  tgat_fused<<<N_DST, 256, 0, stream>>>(
      node_features, edge_features, delta_t, neigh_idx, basis_freq, phase,
      w_qs, ws_bf, ws_bf + ELEMS_PER_MAT,
      fc_w, fc_b, ln_g, ln_b, node_w, node_b, (float*)d_out);
}
